// BasicTransformerBlock_49246095015979
// MI455X (gfx1250) — matrix operand order repaired and checked by reading, not compiled, not run on hardware
//
#include <hip/hip_runtime.h>
#include <hip/hip_bf16.h>

typedef __attribute__((ext_vector_type(16))) __bf16 v16bf;
typedef __attribute__((ext_vector_type(8)))  float  v8f;
typedef __attribute__((ext_vector_type(4)))  unsigned int u32x4;
typedef unsigned long long u64;
typedef unsigned short u16;

union Frag { v16bf v; u32x4 q[2]; };

__device__ __forceinline__ u16 f2bf(float f) {
  unsigned int u = __float_as_uint(f);
  u += 0x7FFFu + ((u >> 16) & 1u);           // round-to-nearest-even (ignores NaN corner)
  return (u16)(u >> 16);
}
__device__ __forceinline__ float bf2f(u16 h) {
  return __uint_as_float(((unsigned int)h) << 16);
}
__device__ __forceinline__ v8f vzero8() {
  v8f z;
#pragma unroll
  for (int i = 0; i < 8; ++i) z[i] = 0.f;
  return z;
}
__device__ __forceinline__ v8f wmma_bf16(v16bf a, v16bf b, v8f c) {
  // 8 args: (neg_a, A, neg_b, B, c_mod, C, reuse_a, reuse_b)
  return __builtin_amdgcn_wmma_f32_16x16x32_bf16(false, a, false, b, (short)0, c, false, false);
}
// byte offset of a __shared__ object within the wave's LDS allocation
__device__ __forceinline__ unsigned lds_off(const void* p) {
  return (unsigned)(unsigned long long)(__attribute__((address_space(3))) const char*)p;
}

// ---------------------------------------------------------------------------
// f32 -> bf16 elementwise convert
// ---------------------------------------------------------------------------
__global__ __launch_bounds__(256) void f2bf_kernel(const float* __restrict__ in,
                                                   u16* __restrict__ out, int n) {
  int i = blockIdx.x * 256 + threadIdx.x;
  if (i < n) out[i] = f2bf(in[i]);
}

// ---------------------------------------------------------------------------
// AdaLN linear: emb[b, 0:2048] = t[b] @ W (1024x2048) + bias
// ---------------------------------------------------------------------------
__global__ __launch_bounds__(256) void adaln_linear_kernel(const float* __restrict__ t,
                                                           const float* __restrict__ w,
                                                           const float* __restrict__ bias,
                                                           float* __restrict__ emb) {
  int n = blockIdx.x * 256 + threadIdx.x;
  int b = blockIdx.y;
  const float* tb = t + (size_t)b * 1024;
  float acc = bias[n];
  for (int d = 0; d < 1024; ++d) acc = fmaf(tb[d], w[(size_t)d * 2048 + n], acc);
  emb[(size_t)b * 2048 + n] = acc;
}

// ---------------------------------------------------------------------------
// AdaLN apply: out_bf16[row] = LN(x[row]) * (1 + scale[b]) + shift[b]
// ---------------------------------------------------------------------------
__global__ __launch_bounds__(256) void adaln_apply_kernel(const float* __restrict__ x,
                                                          const float* __restrict__ emb,
                                                          u16* __restrict__ out, int Srows) {
  __shared__ float rs[256];
  __shared__ float rq[256];
  int row = blockIdx.x, tid = threadIdx.x;
  int b = row / Srows;
  const float* xr = x + (size_t)row * 1024;
  float v[4], s = 0.f, sq = 0.f;
#pragma unroll
  for (int i = 0; i < 4; ++i) {
    v[i] = xr[i * 256 + tid];
    s += v[i];
    sq += v[i] * v[i];
  }
  rs[tid] = s; rq[tid] = sq;
  __syncthreads();
  for (int off = 128; off > 0; off >>= 1) {
    if (tid < off) { rs[tid] += rs[tid + off]; rq[tid] += rq[tid + off]; }
    __syncthreads();
  }
  float mean = rs[0] * (1.f / 1024.f);
  float var  = rq[0] * (1.f / 1024.f) - mean * mean;
  float rstd = rsqrtf(var + 1e-5f);
  const float* eb = emb + (size_t)b * 2048;
#pragma unroll
  for (int i = 0; i < 4; ++i) {
    int n = i * 256 + tid;
    out[(size_t)row * 1024 + n] = f2bf((v[i] - mean) * rstd * (1.f + eb[n]) + eb[1024 + n]);
  }
}

// ---------------------------------------------------------------------------
// GLU gate: g[row,n] = a * gelu_exact(gate), u = [rows, 8192] bf16
// ---------------------------------------------------------------------------
__global__ __launch_bounds__(256) void glu_gate_kernel(const u16* __restrict__ u,
                                                       u16* __restrict__ g, int total) {
  int i = blockIdx.x * 256 + threadIdx.x;
  if (i >= total) return;
  int row = i >> 12, n = i & 4095;
  float a  = bf2f(u[(size_t)row * 8192 + n]);
  float xg = bf2f(u[(size_t)row * 8192 + 4096 + n]);
  float ge = 0.5f * xg * (1.f + erff(xg * 0.70710678118654752f));
  g[i] = f2bf(a * ge);
}

// ---------------------------------------------------------------------------
// WMMA GEMM: C(MxN) = A(MxK,bf16) @ B(KxN,bf16) [+bias] [+residual]
// Block tile 256x64, BK=32, 8 waves; wave w computes rows [w*32, w*32+32)
// (two 16-row A fragments x four 16-col B fragments = 8 WMMA / K-step).
// A tile: row-major LDS (pad 40 u16/row), filled by global_load_async_to_lds
// (ASYNCcnt). B tile: fragment-swizzled LDS (pad 24 u16/lane).
// EPI: 0 = store bf16; 1 = +bias, store bf16; 2 = +bias +residual, store f32
// ---------------------------------------------------------------------------
template <int EPI>
__global__ __launch_bounds__(256) void gemm_bf16_kernel(
    const u16* __restrict__ A, const u16* __restrict__ Bw,
    const float* __restrict__ bias, const float* Res,
    void* Cout, int M, int N, int K) {
  (void)M;
  __shared__ __align__(16) u16 As[256 * 40];      // [row][k] row-major, padded
  __shared__ __align__(16) u16 Bs[4 * 32 * 24];   // [n_tile][lane][elem], padded
  const int tid = threadIdx.x, lane = tid & 31, wave = tid >> 5;
  const int m0 = blockIdx.y * 256, n0 = blockIdx.x * 64;
  const unsigned asBase = lds_off(As);
  v8f acc[2][4];
#pragma unroll
  for (int mi = 0; mi < 2; ++mi)
#pragma unroll
    for (int j = 0; j < 4; ++j) acc[mi][j] = vzero8();

  for (int k0 = 0; k0 < K; k0 += 32) {
    __syncthreads();
    // A tile 256x32: async DMA straight into LDS, 16B per lane per op
#pragma unroll
    for (int it = 0; it < 4; ++it) {
      int i = tid + it * 256;
      int m = i >> 2, c = i & 3;
      u64 ga = (u64)(const void*)(A + (size_t)(m0 + m) * K + k0 + c * 8);
      unsigned la = asBase + (unsigned)(m * 80 + c * 16);
      asm volatile("global_load_async_to_lds_b128 %0, %1, off"
                   :: "v"(la), "v"(ga) : "memory");
    }
    // B tile 32x64: manual fragment swizzle (transpose scatter)
#pragma unroll
    for (int it = 0; it < 2; ++it) {
      int i = tid + it * 256;
      int k = i >> 4, n = (i & 15) << 2;
      const u16* gp = Bw + (size_t)(k0 + k) * N + n0 + n;
      u64 d = *(const u64*)gp;
      if (k0 + 32 < K) __builtin_prefetch(gp + (size_t)32 * N, 0, 1);
      int lh = ((k >> 3) & 1) << 4, e = (k & 7) | ((k >> 4) << 3);
#pragma unroll
      for (int t = 0; t < 4; ++t) {
        int nn = n + t;
        Bs[(((nn >> 4) * 32) + (nn & 15) + lh) * 24 + e] = (u16)(d >> (16 * t));
      }
    }
    asm volatile("s_wait_asynccnt 0x0" ::: "memory");
    __syncthreads();

    Frag a0, a1;
    const u16* ar = As + (wave * 32 + (lane & 15)) * 40 + ((lane >> 4) << 3);
    a0.q[0] = *(const u32x4*)(ar);
    a0.q[1] = *(const u32x4*)(ar + 16);
    a1.q[0] = *(const u32x4*)(ar + 16 * 40);
    a1.q[1] = *(const u32x4*)(ar + 16 * 40 + 16);
#pragma unroll
    for (int j = 0; j < 4; ++j) {
      Frag bf;
      const u16* bp = Bs + (j * 32 + lane) * 24;
      bf.q[0] = *(const u32x4*)(bp);
      bf.q[1] = *(const u32x4*)(bp + 8);
      acc[0][j] = wmma_bf16(a0.v, bf.v, acc[0][j]);
      acc[1][j] = wmma_bf16(a1.v, bf.v, acc[1][j]);
    }
  }
  const int half = lane >> 4, ln = lane & 15;
#pragma unroll
  for (int mi = 0; mi < 2; ++mi) {
#pragma unroll
    for (int j = 0; j < 4; ++j) {
#pragma unroll
      for (int r = 0; r < 8; ++r) {
        int m = m0 + wave * 32 + mi * 16 + r + (half << 3);
        int n = n0 + j * 16 + ln;
        float cv = acc[mi][j][r];
        if (EPI >= 1) cv += bias[n];
        if (EPI == 2) {
          cv += Res[(size_t)m * N + n];
          ((float*)Cout)[(size_t)m * N + n] = cv;
        } else {
          ((u16*)Cout)[(size_t)m * N + n] = f2bf(cv);
        }
      }
    }
  }
}

// ---------------------------------------------------------------------------
// Flash attention. Q/K/V/O layout [B, S, H, 64] bf16. Block = 128 q rows of
// one (b,h); 8 waves, each owns 16 q rows; kv-tile = 32 shared via LDS.
// Per kv-tile per wave: 4 WMMA (QK^T) + 4 WMMA (P·V).
// ---------------------------------------------------------------------------
__global__ __launch_bounds__(256) void flash_kernel(
    const u16* __restrict__ Q, const u16* __restrict__ Kb, const u16* __restrict__ Vb,
    u16* __restrict__ O, int Sq, int Skv, float scale) {
  __shared__ __align__(16) u16 Ks[4 * 32 * 24];   // [(kstep*2+jkv)][lane][elem] padded
  __shared__ __align__(16) u16 Vs[4 * 32 * 24];   // [jdh][lane][elem] padded
  __shared__ __align__(16) u16 Ps[8 * 16 * 40];   // per-wave 16x32 P (row stride 40)
  const int tid = threadIdx.x, lane = tid & 31, wave = tid >> 5;
  const int half = lane >> 4, ln = lane & 15;
  const int h = blockIdx.y, b = blockIdx.z;
  const int qbase = blockIdx.x * 128 + wave * 16;

  // Q fragments for dh 0..31 (aq0) and 32..63 (aq1), loaded straight into
  // A-fragment layout (two contiguous 16B runs per lane per fragment).
  const u16* qp = Q + ((((size_t)b * Sq + qbase + ln) * 16 + h) << 6) + (half << 3);
  Frag aq0, aq1;
  aq0.q[0] = *(const u32x4*)(qp);
  aq0.q[1] = *(const u32x4*)(qp + 16);
  aq1.q[0] = *(const u32x4*)(qp + 32);
  aq1.q[1] = *(const u32x4*)(qp + 48);

  float mrun[8], lrun[8];
  v8f oacc[4];
#pragma unroll
  for (int r = 0; r < 8; ++r) { mrun[r] = -1e30f; lrun[r] = 0.f; }
#pragma unroll
  for (int j = 0; j < 4; ++j) oacc[j] = vzero8();

  u16* Pw = Ps + wave * 640;

  for (int kv0 = 0; kv0 < Skv; kv0 += 32) {
    __syncthreads();
    // cooperative load of K,V tile (32 kv x 64 dh) into fragment-swizzled LDS
#pragma unroll
    for (int it = 0; it < 2; ++it) {
      int i = tid + it * 256;
      int kv = i >> 4, dh = (i & 15) << 2;
      size_t go = (((size_t)b * Skv + kv0 + kv) * 16 + h) << 6;
      u64 dk = *(const u64*)(Kb + go + dh);
      int kdh = dh & 31;
      int dstK = ((((dh >> 5) * 2 + (kv >> 4)) * 32 + (kv & 15) + (((kdh >> 3) & 1) << 4)) * 24) +
                 ((kdh & 7) + ((kdh >> 4) << 3));
      *(u64*)(Ks + dstK) = dk;
      u64 dv = *(const u64*)(Vb + go + dh);
      int lv = (dh & 15) + (((kv >> 3) & 1) << 4);
      int ev = (kv & 7) + ((kv >> 4) << 3);
      int jv = dh >> 4;
#pragma unroll
      for (int t = 0; t < 4; ++t)
        Vs[((jv * 32) + lv + t) * 24 + ev] = (u16)(dv >> (16 * t));
    }
    __syncthreads();

    // scores S = Q K^T  (16 x 32): two n-tiles, two k-steps each
    v8f s0 = vzero8(), s1 = vzero8();
    {
      Frag bk;
      const u16* kp = Ks + lane * 24;                 // (kstep0, j0)
      bk.q[0] = *(const u32x4*)(kp); bk.q[1] = *(const u32x4*)(kp + 8);
      s0 = wmma_bf16(aq0.v, bk.v, s0);
      kp = Ks + (2 * 32 + lane) * 24;                 // (kstep1, j0)
      bk.q[0] = *(const u32x4*)(kp); bk.q[1] = *(const u32x4*)(kp + 8);
      s0 = wmma_bf16(aq1.v, bk.v, s0);
      kp = Ks + (1 * 32 + lane) * 24;                 // (kstep0, j1)
      bk.q[0] = *(const u32x4*)(kp); bk.q[1] = *(const u32x4*)(kp + 8);
      s1 = wmma_bf16(aq0.v, bk.v, s1);
      kp = Ks + (3 * 32 + lane) * 24;                 // (kstep1, j1)
      bk.q[0] = *(const u32x4*)(kp); bk.q[1] = *(const u32x4*)(kp + 8);
      s1 = wmma_bf16(aq1.v, bk.v, s1);
    }

    // online softmax; row stats live redundantly in each 16-lane half
#pragma unroll
    for (int r = 0; r < 8; ++r) {
      float v0 = s0[r] * scale, v1 = s1[r] * scale;
      float mx = fmaxf(v0, v1);
#pragma unroll
      for (int d = 1; d < 16; d <<= 1) mx = fmaxf(mx, __shfl_xor(mx, d, 32));
      float mn = fmaxf(mrun[r], mx);
      float al = __expf(mrun[r] - mn);
      v0 = __expf(v0 - mn);
      v1 = __expf(v1 - mn);
      float rsum = v0 + v1;
#pragma unroll
      for (int d = 1; d < 16; d <<= 1) rsum += __shfl_xor(rsum, d, 32);
      lrun[r] = lrun[r] * al + rsum;
      mrun[r] = mn;
#pragma unroll
      for (int j = 0; j < 4; ++j) oacc[j][r] *= al;
      int m = r + (half << 3);
      Pw[m * 40 + ln]      = f2bf(v0);
      Pw[m * 40 + 16 + ln] = f2bf(v1);
    }
    // wave-local LDS fence before re-reading P in transposed (A-frag) order
    asm volatile("s_wait_dscnt 0" ::: "memory");
    __builtin_amdgcn_wave_barrier();

    Frag ap;
    const u16* pp = Pw + ln * 40 + (half << 3);
    ap.q[0] = *(const u32x4*)(pp);
    ap.q[1] = *(const u32x4*)(pp + 16);
#pragma unroll
    for (int j = 0; j < 4; ++j) {
      Frag bv;
      const u16* vp = Vs + (j * 32 + lane) * 24;
      bv.q[0] = *(const u32x4*)(vp); bv.q[1] = *(const u32x4*)(vp + 8);
      oacc[j] = wmma_bf16(ap.v, bv.v, oacc[j]);
    }
  }

#pragma unroll
  for (int r = 0; r < 8; ++r) {
    float inv = 1.f / lrun[r];
    int m = qbase + r + (half << 3);
    size_t ro = (((size_t)b * Sq + m) * 16 + h) << 6;
#pragma unroll
    for (int j = 0; j < 4; ++j)
      O[ro + j * 16 + ln] = f2bf(oacc[j][r] * inv);
  }
}

// ---------------------------------------------------------------------------
// Host-side orchestration
// ---------------------------------------------------------------------------
extern "C" void kernel_launch(void* const* d_in, const int* in_sizes, int n_in,
                              void* d_out, int out_size, void* d_ws, size_t ws_size,
                              hipStream_t stream) {
  (void)in_sizes; (void)n_in; (void)out_size; (void)ws_size;
  const float* x    = (const float*)d_in[0];
  const float* t    = (const float*)d_in[1];
  const float* ctx  = (const float*)d_in[2];
  const float* a1wq = (const float*)d_in[3];
  const float* a1wk = (const float*)d_in[4];
  const float* a1wv = (const float*)d_in[5];
  const float* a1wo = (const float*)d_in[6];
  const float* a1bo = (const float*)d_in[7];
  const float* a2wq = (const float*)d_in[8];
  const float* a2wk = (const float*)d_in[9];
  const float* a2wv = (const float*)d_in[10];
  const float* a2wo = (const float*)d_in[11];
  const float* a2bo = (const float*)d_in[12];
  const float* fw1f = (const float*)d_in[13];
  const float* fb1  = (const float*)d_in[14];
  const float* fw2f = (const float*)d_in[15];
  const float* fb2  = (const float*)d_in[16];
  const float* n1w  = (const float*)d_in[17];
  const float* n1b  = (const float*)d_in[18];
  const float* n2w  = (const float*)d_in[19];
  const float* n2b  = (const float*)d_in[20];
  const float* n3w  = (const float*)d_in[21];
  const float* n3b  = (const float*)d_in[22];
  float* out = (float*)d_out;

  const int B = 2, S = 2048, CTX = 256, D = 1024, H = 16;
  const int M = B * S;            // 4096 rows
  const int MC = B * CTX;         // 512 context rows

  char* p = (char*)d_ws;
  auto alloc = [&](size_t bytes) -> char* {
    char* r = p;
    p += (bytes + 255) & ~(size_t)255;
    return r;
  };
  // bf16 weights / context
  u16* wq1  = (u16*)alloc((size_t)D * D * 2);
  u16* wk1  = (u16*)alloc((size_t)D * D * 2);
  u16* wv1  = (u16*)alloc((size_t)D * D * 2);
  u16* wo1  = (u16*)alloc((size_t)D * D * 2);
  u16* wq2  = (u16*)alloc((size_t)D * D * 2);
  u16* wk2  = (u16*)alloc((size_t)D * D * 2);
  u16* wv2  = (u16*)alloc((size_t)D * D * 2);
  u16* wo2  = (u16*)alloc((size_t)D * D * 2);
  u16* fw1  = (u16*)alloc((size_t)D * 8192 * 2);
  u16* fw2  = (u16*)alloc((size_t)4096 * D * 2);
  u16* ctxb = (u16*)alloc((size_t)MC * D * 2);
  // activations
  float* emb1 = (float*)alloc((size_t)B * 2048 * 4);
  float* emb2 = (float*)alloc((size_t)B * 2048 * 4);
  float* emb3 = (float*)alloc((size_t)B * 2048 * 4);
  u16* hb = (u16*)alloc((size_t)M * D * 2);
  u16* qb = (u16*)alloc((size_t)M * D * 2);
  u16* kb = (u16*)alloc((size_t)M * D * 2);
  u16* vb = (u16*)alloc((size_t)M * D * 2);
  u16* ob = (u16*)alloc((size_t)M * D * 2);
  float* x2 = (float*)alloc((size_t)M * D * 4);
  u16* ub = (u16*)alloc((size_t)1024 * 8192 * 2);
  u16* gb = (u16*)alloc((size_t)1024 * 4096 * 2);

  auto conv = [&](const float* src, u16* dst, int n) {
    f2bf_kernel<<<dim3((n + 255) / 256), dim3(256), 0, stream>>>(src, dst, n);
  };
  conv(a1wq, wq1, D * D); conv(a1wk, wk1, D * D); conv(a1wv, wv1, D * D); conv(a1wo, wo1, D * D);
  conv(a2wq, wq2, D * D); conv(a2wk, wk2, D * D); conv(a2wv, wv2, D * D); conv(a2wo, wo2, D * D);
  conv(fw1f, fw1, D * 8192);
  conv(fw2f, fw2, 4096 * D);
  conv(ctx, ctxb, MC * D);

  dim3 blk(256);
  dim3 gemmDD(D / 64, M / 256);       // 16 x 16

  // ---- stage 1: self-attention ----
  adaln_linear_kernel<<<dim3(8, B), blk, 0, stream>>>(t, n1w, n1b, emb1);
  adaln_apply_kernel<<<dim3(M), blk, 0, stream>>>(x, emb1, hb, S);
  gemm_bf16_kernel<0><<<gemmDD, blk, 0, stream>>>(hb, wq1, nullptr, nullptr, qb, M, D, D);
  gemm_bf16_kernel<0><<<gemmDD, blk, 0, stream>>>(hb, wk1, nullptr, nullptr, kb, M, D, D);
  gemm_bf16_kernel<0><<<gemmDD, blk, 0, stream>>>(hb, wv1, nullptr, nullptr, vb, M, D, D);
  flash_kernel<<<dim3(S / 128, H, B), blk, 0, stream>>>(qb, kb, vb, ob, S, S, 0.125f);
  gemm_bf16_kernel<2><<<gemmDD, blk, 0, stream>>>(ob, wo1, a1bo, x, x2, M, D, D);

  // ---- stage 2: cross-attention ----
  adaln_linear_kernel<<<dim3(8, B), blk, 0, stream>>>(t, n2w, n2b, emb2);
  adaln_apply_kernel<<<dim3(M), blk, 0, stream>>>(x2, emb2, hb, S);
  gemm_bf16_kernel<0><<<gemmDD, blk, 0, stream>>>(hb, wq2, nullptr, nullptr, qb, M, D, D);
  gemm_bf16_kernel<0><<<dim3(D / 64, MC / 256), blk, 0, stream>>>(ctxb, wk2, nullptr, nullptr, kb, MC, D, D);
  gemm_bf16_kernel<0><<<dim3(D / 64, MC / 256), blk, 0, stream>>>(ctxb, wv2, nullptr, nullptr, vb, MC, D, D);
  flash_kernel<<<dim3(S / 128, H, B), blk, 0, stream>>>(qb, kb, vb, ob, S, CTX, 0.125f);
  gemm_bf16_kernel<2><<<gemmDD, blk, 0, stream>>>(ob, wo2, a2bo, x2, x2, M, D, D);

  // ---- stage 3: gated FFN (chunked over 1024-row slabs) ----
  adaln_linear_kernel<<<dim3(8, B), blk, 0, stream>>>(t, n3w, n3b, emb3);
  adaln_apply_kernel<<<dim3(M), blk, 0, stream>>>(x2, emb3, hb, S);
  for (int c = 0; c < 4; ++c) {
    const u16* hc = hb + (size_t)c * 1024 * D;
    gemm_bf16_kernel<1><<<dim3(8192 / 64, 1024 / 256), blk, 0, stream>>>(
        hc, fw1, fb1, nullptr, ub, 1024, 8192, D);
    glu_gate_kernel<<<dim3((1024 * 4096) / 256), blk, 0, stream>>>(ub, gb, 1024 * 4096);
    gemm_bf16_kernel<2><<<dim3(D / 64, 1024 / 256), blk, 0, stream>>>(
        gb, fw2, fb2, x2 + (size_t)c * 1024 * D, out + (size_t)c * 1024 * D, 1024, D, 4096);
  }
}
